// MultiHeadSelfAttention_57784490001186
// MI455X (gfx1250) — matrix operand order repaired and checked by reading, not compiled, not run on hardware
//
#include <hip/hip_runtime.h>

// ---------------------------------------------------------------------------
// CDNA5 (gfx1250) multi-head self-attention, wave32 + v_wmma_f32_16x16x32_bf16
// S=4096, E=1024, H=16, D=64
// ---------------------------------------------------------------------------

#define S_LEN 4096
#define E_DIM 1024
#define N_HEADS 16
#define HD 64

typedef __bf16 v16bf __attribute__((ext_vector_type(16)));
typedef float  v8f   __attribute__((ext_vector_type(8)));

__device__ __forceinline__ v8f wmma_bf16(v16bf a, v16bf b, v8f c) {
  // 8 args: (neg_a, A, neg_b, B, c_mod, C, reuse_a, reuse_b)
  return __builtin_amdgcn_wmma_f32_16x16x32_bf16(false, a, false, b, (short)0, c,
                                                 false, false);
}

// A-fragment (16x32 bf16, M x K). Lane l: m = l&15, k-base = (l>>4)*8.
// VGPR j (0..3): k = kb+2j, kb+2j+1 ; VGPR 4..7: k = 16+kb+2(j-4), +1
// Per lane this is two contiguous 16-byte runs -> two global_load_b128.
__device__ __forceinline__ v16bf load_a_frag(const __bf16* __restrict__ src, int ld,
                                             int row0, int k0, int lane) {
  int m  = lane & 15;
  int kb = (lane >> 4) * 8;
  const __bf16* p = src + (size_t)(row0 + m) * ld + k0;
  v16bf a;
#pragma unroll
  for (int j = 0; j < 4; ++j) {
    a[2 * j]         = p[kb + 2 * j];
    a[2 * j + 1]     = p[kb + 2 * j + 1];
    a[8 + 2 * j]     = p[16 + kb + 2 * j];
    a[8 + 2 * j + 1] = p[16 + kb + 2 * j + 1];
  }
  return a;
}

// B-fragment (32x16 bf16, K x N) gathered from a row-major matrix `src` whose
// ROW index is N and COLUMN index is K (i.e. B = src^T). Lane l: n = l&15,
// k-base = (l>>4)*8, VGPR j (0..3): k = kb + 2j, +1; VGPR 4..7: k = 16+kb+2(j-4), +1.
// Two contiguous 16-byte runs per lane -> two global_load_b128.
__device__ __forceinline__ v16bf load_bT_frag(const __bf16* __restrict__ src, int ld,
                                              int n0, int k0, int lane) {
  int n  = n0 + (lane & 15);
  int kb = (lane >> 4) * 8;
  const __bf16* p = src + (size_t)n * ld + k0 + kb;
  v16bf b;
#pragma unroll
  for (int j = 0; j < 4; ++j) {
    b[2 * j]         = p[2 * j];
    b[2 * j + 1]     = p[2 * j + 1];
    b[8 + 2 * j]     = p[16 + 2 * j];
    b[8 + 2 * j + 1] = p[16 + 2 * j + 1];
  }
  return b;
}

// ---------------------------------------------------------------------------
// fp32 -> bf16 downcast
// ---------------------------------------------------------------------------
__global__ void __launch_bounds__(256) cvt_bf16_kernel(const float* __restrict__ src,
                                                       __bf16* __restrict__ dst, int n) {
  int i = blockIdx.x * blockDim.x + threadIdx.x;
  if (i < n) dst[i] = (__bf16)src[i];
}

// ---------------------------------------------------------------------------
// Fused QKV projection: blockIdx.z selects Q/K/V. Each wave computes a
// 32(row) x 64(col) tile (2 M-tiles x 4 N-tiles): every weight B-fragment is
// reused by two WMMAs, doubling math per fetched weight byte.
// Q is scaled by 1/sqrt(D) at store. Q,K -> [H][S][D]; V -> transposed [H][D][S].
// ---------------------------------------------------------------------------
__global__ void __launch_bounds__(128) qkv_proj_kernel(
    const __bf16* __restrict__ xb,
    const __bf16* __restrict__ wq, const __bf16* __restrict__ wk,
    const __bf16* __restrict__ wv,
    const float* __restrict__ bq, const float* __restrict__ bk,
    const float* __restrict__ bv,
    __bf16* __restrict__ Qh, __bf16* __restrict__ Kh, __bf16* __restrict__ Vt) {
  int lane = threadIdx.x & 31;
  int wave = threadIdx.x >> 5;
  int s0   = blockIdx.x * 128 + wave * 32;  // 32 rows per wave
  int h    = blockIdx.y;
  int e0   = h * HD;
  int mode = blockIdx.z;  // 0=Q 1=K 2=V

  const __bf16* w    = (mode == 0) ? wq : (mode == 1) ? wk : wv;
  const float*  bias = (mode == 0) ? bq : (mode == 1) ? bk : bv;

  v8f acc[2][4] = {};
  for (int k0 = 0; k0 < E_DIM; k0 += 32) {
    v16bf a0 = load_a_frag(xb, E_DIM, s0, k0, lane);
    v16bf a1 = load_a_frag(xb, E_DIM, s0 + 16, k0, lane);
#pragma unroll
    for (int nt = 0; nt < 4; ++nt) {
      v16bf b = load_bT_frag(w, E_DIM, e0 + nt * 16, k0, lane);  // row of W = out ch
      acc[0][nt] = wmma_bf16(a0, b, acc[0][nt]);
      acc[1][nt] = wmma_bf16(a1, b, acc[1][nt]);
    }
  }

  int n    = lane & 15;
  int mrow = (lane >> 4) * 8;
  float scale = (mode == 0) ? 0.125f : 1.0f;  // 1/sqrt(64) folded into Q
#pragma unroll
  for (int mt = 0; mt < 2; ++mt) {
#pragma unroll
    for (int nt = 0; nt < 4; ++nt) {
      int d = nt * 16 + n;  // 0..63 within head
      float bb = bias[e0 + d];
#pragma unroll
      for (int j = 0; j < 8; ++j) {
        int s = s0 + mt * 16 + mrow + j;
        __bf16 val = (__bf16)((acc[mt][nt][j] + bb) * scale);
        if (mode == 2)
          Vt[(size_t)h * HD * S_LEN + (size_t)d * S_LEN + s] = val;
        else if (mode == 0)
          Qh[(size_t)h * S_LEN * HD + (size_t)s * HD + d] = val;
        else
          Kh[(size_t)h * S_LEN * HD + (size_t)s * HD + d] = val;
      }
    }
  }
}

// ---------------------------------------------------------------------------
// Flash attention: block = 4 waves = 64 queries of one head.
// Online softmax over key tiles of 64; P transposed via per-wave LDS tile.
// ---------------------------------------------------------------------------
__global__ void __launch_bounds__(128) flash_attn_kernel(
    const __bf16* __restrict__ Qh, const __bf16* __restrict__ Kh,
    const __bf16* __restrict__ Vt, __bf16* __restrict__ attn) {
  __shared__ __bf16 ldsP[4][16 * 64];

  int lane = threadIdx.x & 31;
  int wave = threadIdx.x >> 5;
  int h    = blockIdx.y;
  int q0   = blockIdx.x * 64 + wave * 16;

  const __bf16* Q = Qh + (size_t)h * S_LEN * HD;
  const __bf16* K = Kh + (size_t)h * S_LEN * HD;
  const __bf16* V = Vt + (size_t)h * HD * S_LEN;

  // Q fragments for the whole K-dim (D=64 -> two k-steps), kept resident
  v16bf aq0 = load_a_frag(Q, HD, q0, 0, lane);
  v16bf aq1 = load_a_frag(Q, HD, q0, 32, lane);

  float mi[8], li[8];
#pragma unroll
  for (int j = 0; j < 8; ++j) { mi[j] = -3.0e38f; li[j] = 0.0f; }
  v8f o[4] = {};

  for (int t0 = 0; t0 < S_LEN; t0 += 64) {
    // --- scores: 16 x 64 tile, 8 WMMA ---
    v8f sc[4];
#pragma unroll
    for (int nt = 0; nt < 4; ++nt) {
      v8f c = {};
      v16bf b0 = load_bT_frag(K, HD, t0 + nt * 16, 0, lane);
      c = wmma_bf16(aq0, b0, c);
      v16bf b1 = load_bT_frag(K, HD, t0 + nt * 16, 32, lane);
      c = wmma_bf16(aq1, b1, c);
      sc[nt] = c;
    }

    // --- online softmax: row max / rescale (16-wide shfl within wave32) ---
    float nm[8], alpha[8];
#pragma unroll
    for (int j = 0; j < 8; ++j) {
      float v = fmaxf(fmaxf(sc[0][j], sc[1][j]), fmaxf(sc[2][j], sc[3][j]));
      v = fmaxf(v, __shfl_xor(v, 1, 32));
      v = fmaxf(v, __shfl_xor(v, 2, 32));
      v = fmaxf(v, __shfl_xor(v, 4, 32));
      v = fmaxf(v, __shfl_xor(v, 8, 32));
      nm[j]    = fmaxf(mi[j], v);
      alpha[j] = __expf(mi[j] - nm[j]);
      mi[j]    = nm[j];
    }
#pragma unroll
    for (int nt = 0; nt < 4; ++nt)
#pragma unroll
      for (int j = 0; j < 8; ++j) sc[nt][j] = __expf(sc[nt][j] - nm[j]);
#pragma unroll
    for (int j = 0; j < 8; ++j) {
      float rs = sc[0][j] + sc[1][j] + sc[2][j] + sc[3][j];
      rs += __shfl_xor(rs, 1, 32);
      rs += __shfl_xor(rs, 2, 32);
      rs += __shfl_xor(rs, 4, 32);
      rs += __shfl_xor(rs, 8, 32);
      li[j] = li[j] * alpha[j] + rs;
#pragma unroll
      for (int dt = 0; dt < 4; ++dt) o[dt][j] *= alpha[j];
    }

    // --- C-layout -> A-layout transpose of P through per-wave LDS tile ---
    {
      int n = lane & 15, mrow = (lane >> 4) * 8;
#pragma unroll
      for (int nt = 0; nt < 4; ++nt)
#pragma unroll
        for (int j = 0; j < 8; ++j)
          ldsP[wave][(mrow + j) * 64 + nt * 16 + n] = (__bf16)sc[nt][j];
    }
    // same-wave DS RAW: DS ops are in-order; fence the compiler and counter
    asm volatile("s_wait_dscnt 0" ::: "memory");

    v16bf ap0 = load_a_frag(&ldsP[wave][0], 64, 0, 0, lane);
    v16bf ap1 = load_a_frag(&ldsP[wave][0], 64, 0, 32, lane);

    // --- O += P @ V  (V stored transposed [D][S] -> contiguous B reads) ---
#pragma unroll
    for (int dt = 0; dt < 4; ++dt) {
      v16bf b0 = load_bT_frag(V, S_LEN, dt * 16, t0, lane);
      o[dt] = wmma_bf16(ap0, b0, o[dt]);
      v16bf b1 = load_bT_frag(V, S_LEN, dt * 16, t0 + 32, lane);
      o[dt] = wmma_bf16(ap1, b1, o[dt]);
    }
  }

  // --- normalize and emit bf16 attention output [S][E], e = h*64 + d ---
  int n = lane & 15, mrow = (lane >> 4) * 8;
#pragma unroll
  for (int dt = 0; dt < 4; ++dt)
#pragma unroll
    for (int j = 0; j < 8; ++j) {
      int s = q0 + mrow + j;
      int e = h * HD + dt * 16 + n;
      attn[(size_t)s * E_DIM + e] = (__bf16)(o[dt][j] / li[j]);
    }
}

// ---------------------------------------------------------------------------
// Output projection: out = attn @ Wo^T + bo, fp32 result.
// Same 32x64-per-wave register blocking as the QKV GEMM.
// ---------------------------------------------------------------------------
__global__ void __launch_bounds__(128) out_proj_kernel(
    const __bf16* __restrict__ attn, const __bf16* __restrict__ wo,
    const float* __restrict__ bo, float* __restrict__ out) {
  int lane = threadIdx.x & 31;
  int wave = threadIdx.x >> 5;
  int s0   = blockIdx.x * 128 + wave * 32;
  int e0   = blockIdx.y * 64;

  v8f acc[2][4] = {};
  for (int k0 = 0; k0 < E_DIM; k0 += 32) {
    v16bf a0 = load_a_frag(attn, E_DIM, s0, k0, lane);
    v16bf a1 = load_a_frag(attn, E_DIM, s0 + 16, k0, lane);
#pragma unroll
    for (int nt = 0; nt < 4; ++nt) {
      v16bf b = load_bT_frag(wo, E_DIM, e0 + nt * 16, k0, lane);
      acc[0][nt] = wmma_bf16(a0, b, acc[0][nt]);
      acc[1][nt] = wmma_bf16(a1, b, acc[1][nt]);
    }
  }

  int n = lane & 15, mrow = (lane >> 4) * 8;
#pragma unroll
  for (int mt = 0; mt < 2; ++mt) {
#pragma unroll
    for (int nt = 0; nt < 4; ++nt) {
      float bb = bo[e0 + nt * 16 + n];
#pragma unroll
      for (int j = 0; j < 8; ++j)
        out[(size_t)(s0 + mt * 16 + mrow + j) * E_DIM + e0 + nt * 16 + n] =
            acc[mt][nt][j] + bb;
    }
  }
}

// ---------------------------------------------------------------------------
// launch
// ---------------------------------------------------------------------------
extern "C" void kernel_launch(void* const* d_in, const int* in_sizes, int n_in,
                              void* d_out, int out_size, void* d_ws, size_t ws_size,
                              hipStream_t stream) {
  const float* x  = (const float*)d_in[0];
  const float* Wq = (const float*)d_in[1];
  const float* bq = (const float*)d_in[2];
  const float* Wk = (const float*)d_in[3];
  const float* bk = (const float*)d_in[4];
  const float* Wv = (const float*)d_in[5];
  const float* bv = (const float*)d_in[6];
  const float* Wo = (const float*)d_in[7];
  const float* bo = (const float*)d_in[8];
  float* out = (float*)d_out;

  const size_t SE  = (size_t)S_LEN * E_DIM;         // 4M elems
  const size_t EE  = (size_t)E_DIM * E_DIM;         // 1M elems
  const size_t HSD = (size_t)N_HEADS * S_LEN * HD;  // 4M elems

  __bf16* xb   = (__bf16*)d_ws;
  __bf16* wqb  = xb + SE;
  __bf16* wkb  = wqb + EE;
  __bf16* wvb  = wkb + EE;
  __bf16* wob  = wvb + EE;
  __bf16* Qh   = wob + EE;   // [H][S][D]
  __bf16* Kh   = Qh + HSD;   // [H][S][D]
  __bf16* Vt   = Kh + HSD;   // [H][D][S]
  __bf16* attn = Vt + HSD;   // [S][E]

  cvt_bf16_kernel<<<(int)(SE / 256), 256, 0, stream>>>(x, xb, (int)SE);
  cvt_bf16_kernel<<<(int)(EE / 256), 256, 0, stream>>>(Wq, wqb, (int)EE);
  cvt_bf16_kernel<<<(int)(EE / 256), 256, 0, stream>>>(Wk, wkb, (int)EE);
  cvt_bf16_kernel<<<(int)(EE / 256), 256, 0, stream>>>(Wv, wvb, (int)EE);
  cvt_bf16_kernel<<<(int)(EE / 256), 256, 0, stream>>>(Wo, wob, (int)EE);

  dim3 gq(S_LEN / 128, N_HEADS, 3);
  qkv_proj_kernel<<<gq, 128, 0, stream>>>(xb, wqb, wkb, wvb, bq, bk, bv, Qh, Kh, Vt);

  dim3 ga(S_LEN / 64, N_HEADS);
  flash_attn_kernel<<<ga, 128, 0, stream>>>(Qh, Kh, Vt, attn);

  dim3 go(S_LEN / 128, E_DIM / 64);
  out_proj_kernel<<<go, 128, 0, stream>>>(attn, wob, bo, out);
}
